// MultiScaleVQVAE2_42339787604563
// MI455X (gfx1250) — matrix operand order repaired and checked by reading, not compiled, not run on hardware
//
#include <hip/hip_runtime.h>
#include <stdint.h>

// ---------------------------------------------------------------------------
// CDNA5 (gfx1250) WMMA types and helpers
// ---------------------------------------------------------------------------
typedef __attribute__((ext_vector_type(16))) __bf16 v16bf;
typedef __attribute__((ext_vector_type(8)))  float  v8f;

// Round-to-nearest-even fp32->bf16: one-time weight/codebook passes.
__device__ __forceinline__ __bf16 f2bf(float f) {
  union { float f; uint32_t u; } v; v.f = f;
  uint32_t r = (v.u + 0x7FFFu + ((v.u >> 16) & 1u)) >> 16;
  union { uint16_t u; __bf16 b; } o; o.u = (uint16_t)r;
  return o.b;
}

// Truncating fp32->bf16: hot A-operand path (1 VALU op: take high half).
__device__ __forceinline__ __bf16 f2bf_trunc(float f) {
  union { float f; uint32_t u; } v; v.f = f;
  union { uint16_t u; __bf16 b; } o; o.u = (uint16_t)(v.u >> 16);
  return o.b;
}

#define ACT_NONE 0
#define ACT_RELU 1
#define ACT_SIGM 2

// ---------------------------------------------------------------------------
// FAST implicit-GEMM 3D conv via V_WMMA_F32_16X16X32_BF16, NT column tiles
// per wave (one A fragment feeds NT independent WMMAs; B fragments are
// contiguous aligned 32B vector loads from pre-reordered bf16 weights
// [Cout][tap][CIN]).  K-order (tap, cin), CIN % 32 == 0, so tap
// decomposition/bounds/spatial offset are computed once per tap.
// (nb,dzo,ho) are wave-uniform (Wout%16==0, M%16==0) -> readfirstlane tap
// skip is scalar and EXEC stays all-ones at every WMMA (ISA 7.12).
// ---------------------------------------------------------------------------
template <int KS, int STR, int PAD, int LDIL, int CIN, int DIN, int HIN, int WIN, int NT>
__global__ void conv3d_wmma_fast(const float* __restrict__ x,
                                 const __bf16* __restrict__ wr,
                                 const float* __restrict__ bias,
                                 float* __restrict__ y,
                                 int N, int Cout, int Dout, int Hout, int Wout,
                                 int act) {
  constexpr int  K2   = KS * KS;
  constexpr int  KV   = KS * K2;
  constexpr int  KTOT = CIN * KV;
  constexpr long HW   = (long)HIN * WIN;
  constexpr long CS   = (long)DIN * HW;

  const int lane = threadIdx.x;
  const int half = lane >> 4;
  const int l15  = lane & 15;

  const int M   = N * Dout * Hout * Wout;
  const int row = blockIdx.x * 16 + l15;
  const int rr  = row < M ? row : M - 1;
  int wo = rr % Wout; int t = rr / Wout;
  int ho = t % Hout;  t /= Hout;
  int dzo = t % Dout; int nb = t / Dout;
  const bool rowOk = row < M;

  const int colBase = blockIdx.y * (16 * NT);
  const __bf16* wcol[NT];
#pragma unroll
  for (int j = 0; j < NT; ++j) {
    const int cj = colBase + j * 16 + l15;
    wcol[j] = wr + (long)(cj < Cout ? cj : 0) * KTOT + half * 16;
  }
  const long rowbase = (long)nb * CIN * CS + (long)half * 16 * CS;

  v8f acc[NT] = {};
  for (int tap = 0; tap < KV; ++tap) {
    const int kd = tap / K2;
    const int kh = (tap / KS) % KS;
    const int kw = tap % KS;
    const int id = dzo * STR + kd - PAD;
    const int ih = ho  * STR + kh - PAD;
    const int iw = wo  * STR + kw - PAD;
    int sd, sh2, sw2; bool dOk, hOk, wOk;
    if (LDIL == 2) {
      dOk = (id >= 0) && ((id & 1) == 0) && ((id >> 1) < DIN); sd  = id >> 1;
      hOk = (ih >= 0) && ((ih & 1) == 0) && ((ih >> 1) < HIN); sh2 = ih >> 1;
      wOk = (iw >= 0) && ((iw & 1) == 0) && ((iw >> 1) < WIN); sw2 = iw >> 1;
    } else {
      dOk = (unsigned)id < (unsigned)DIN; sd  = id;
      hOk = (unsigned)ih < (unsigned)HIN; sh2 = ih;
      wOk = (unsigned)iw < (unsigned)WIN; sw2 = iw;
    }
    if (!__builtin_amdgcn_readfirstlane((int)(dOk && hOk))) continue;  // scalar skip

    const bool aOk = rowOk && wOk && dOk && hOk;
    // branch-free predication: always load from a valid address, zero by select
    const float* xs = aOk ? (x + rowbase + (long)sd * HW + (long)sh2 * WIN + sw2)
                          : (x + (long)half * 16 * CS);
    const long wtap = (long)tap * CIN;
    __builtin_prefetch(wcol[0] + wtap + CIN, 0, 0);   // next tap's weight stream

    for (int c0 = 0; c0 < CIN; c0 += 32) {
      const float* xc = xs + (long)c0 * CS;
      v16bf a;
#pragma unroll
      for (int i = 0; i < 16; ++i) {
        const int ci = i;     // K offset within step (A layout: k = 16*half + i)
        const float raw = xc[(long)ci * CS];
        a[i] = f2bf_trunc(aOk ? raw : 0.f);
      }
#pragma unroll
      for (int j = 0; j < NT; ++j) {
        const v16bf b = *(const v16bf*)(wcol[j] + wtap + c0);
        acc[j] = __builtin_amdgcn_wmma_f32_16x16x32_bf16(false, a, false, b, (short)0,
                                                         acc[j], false, false);
      }
    }
  }

  float bb[NT];
#pragma unroll
  for (int j = 0; j < NT; ++j) {
    const int cj = colBase + j * 16 + l15;
    bb[j] = (bias && cj < Cout) ? bias[cj] : 0.f;
  }
  const long ohw = (long)Hout * Wout;
  const long cstr = (long)Dout * ohw;
#pragma unroll
  for (int r = 0; r < 8; ++r) {
    const int m = blockIdx.x * 16 + r + 8 * half;
    if (m >= M) continue;
    int wo2 = m % Wout; int t2 = m / Wout;
    int ho2 = t2 % Hout; t2 /= Hout;
    int do2 = t2 % Dout; int nb2 = t2 / Dout;
    const long base = ((long)nb2 * Cout) * cstr + (long)do2 * ohw + (long)ho2 * Wout + wo2;
#pragma unroll
    for (int j = 0; j < NT; ++j) {
      const int cj = colBase + j * 16 + l15;
      if (cj < Cout) {
        float v = acc[j][r] + bb[j];
        if (act == ACT_RELU) v = fmaxf(v, 0.f);
        else if (act == ACT_SIGM) v = 1.f / (1.f + __expf(-v));
        y[base + (long)cj * cstr] = v;
      }
    }
  }
}

// ---------------------------------------------------------------------------
// First conv (x[4,3,16,128,128] -> [4,64,8,64,64], k4 s2 p1 + ReLU).
// Cin padded 3->4: K = tap*4 + cin, Ktot = 256 (shifts/masks only).
// Branch-free predicated A loads (clamped address + value select) so the 16
// loads issue as one clause with a single wait. Cout=64 = 4 column tiles.
// ---------------------------------------------------------------------------
__global__ void conv3d_wmma_first(const float* __restrict__ x,
                                  const __bf16* __restrict__ wr,
                                  const float* __restrict__ bias,
                                  float* __restrict__ y) {
  constexpr int  DIN = 16, HIN = 128, WIN = 128;
  constexpr long HW = (long)HIN * WIN;
  constexpr long CS = (long)DIN * HW;
  constexpr int  KTOT = 256;

  const int lane = threadIdx.x;
  const int half = lane >> 4;
  const int l15  = lane & 15;

  const int row = blockIdx.x * 16 + l15;        // M = 131072, grid exact
  const int wo  = row & 63;
  const int ho  = (row >> 6) & 63;
  const int dzo = (row >> 12) & 7;
  const int nb  = row >> 15;

  const __bf16* wcol[4];
#pragma unroll
  for (int j = 0; j < 4; ++j)
    wcol[j] = wr + (long)(j * 16 + l15) * KTOT + half * 16;
  const long rowbase = (long)nb * 3 * CS;

  v8f acc[4] = {};
  for (int k0 = 0; k0 < KTOT; k0 += 32) {
    v16bf a;
#pragma unroll
    for (int i = 0; i < 16; ++i) {
      const int ka  = k0 + half * 16 + i;
      const int tap = ka >> 2;
      const int cin = ka & 3;
      const int kd = tap >> 4, kh = (tap >> 2) & 3, kw = tap & 3;
      const int id = (dzo << 1) + kd - 1;
      const int ih = (ho  << 1) + kh - 1;
      const int iw = (wo  << 1) + kw - 1;
      const bool ok = (cin < 3) && ((unsigned)id < (unsigned)DIN) &&
                      ((unsigned)ih < (unsigned)HIN) && ((unsigned)iw < (unsigned)WIN);
      const long idx = ok ? (rowbase + (long)cin * CS + (long)id * HW + (long)ih * WIN + iw)
                          : rowbase;              // always-valid fallback
      const float raw = x[idx];
      a[i] = f2bf_trunc(ok ? raw : 0.f);
    }
#pragma unroll
    for (int j = 0; j < 4; ++j) {
      const v16bf b = *(const v16bf*)(wcol[j] + k0);
      acc[j] = __builtin_amdgcn_wmma_f32_16x16x32_bf16(false, a, false, b, (short)0,
                                                       acc[j], false, false);
    }
  }

#pragma unroll
  for (int r = 0; r < 8; ++r) {
    const int m = blockIdx.x * 16 + r + 8 * half;
    const int wo2 = m & 63, ho2 = (m >> 6) & 63, do2 = (m >> 12) & 7, nb2 = m >> 15;
    const long base = (long)nb2 * 64 * 32768 + (long)do2 * 4096 + ho2 * 64 + wo2;
#pragma unroll
    for (int j = 0; j < 4; ++j) {
      const int cj = j * 16 + l15;
      y[base + (long)cj * 32768] = fmaxf(acc[j][r] + bias[cj], 0.f);   // ReLU
    }
  }
}

// ---------------------------------------------------------------------------
// VQ distance GEMM: dots[r][k] = sum_c xf[r][c]*emb[k][c] (bf16 WMMA, NT tiles)
// ---------------------------------------------------------------------------
template <int C, int NT>
__global__ void vq_dots_fast(const float* __restrict__ x, const __bf16* __restrict__ ebf,
                             float* __restrict__ dots, int B, int D, int H, int W, int K) {
  const int lane = threadIdx.x;
  const int half = lane >> 4;
  const int l15  = lane & 15;

  const int M = B * D * H * W;
  const int row = blockIdx.x * 16 + l15;
  const int rr  = row < M ? row : M - 1;
  int w = rr % W; int t = rr / W;
  int h = t % H;  t /= H;
  int d = t % D;  int b = t / D;
  const long spat = (long)D * H * W;
  const bool rowOk = row < M;
  const float* xs = rowOk ? (x + (long)b * C * spat + (long)d * H * W + (long)h * W + w
                               + (long)half * 16 * spat)
                          : (x + (long)half * 16 * spat);

  const int colBase = blockIdx.y * (16 * NT);
  const __bf16* es[NT];
#pragma unroll
  for (int j = 0; j < NT; ++j) {
    const int cj = colBase + j * 16 + l15;
    es[j] = ebf + (long)(cj < K ? cj : 0) * C + half * 16;
  }

  v8f acc[NT] = {};
  for (int c0 = 0; c0 < C; c0 += 32) {
    const float* xc = xs + (long)c0 * spat;
    v16bf a;
#pragma unroll
    for (int i = 0; i < 16; ++i) {
      const int ci = i;
      const float raw = xc[(long)ci * spat];
      a[i] = f2bf_trunc(rowOk ? raw : 0.f);
    }
#pragma unroll
    for (int j = 0; j < NT; ++j) {
      const v16bf bm = *(const v16bf*)(es[j] + c0);
      acc[j] = __builtin_amdgcn_wmma_f32_16x16x32_bf16(false, a, false, bm, (short)0,
                                                       acc[j], false, false);
    }
  }
#pragma unroll
  for (int r = 0; r < 8; ++r) {
    const int m = blockIdx.x * 16 + r + 8 * half;
    if (m >= M) continue;
#pragma unroll
    for (int j = 0; j < NT; ++j) {
      const int cj = colBase + j * 16 + l15;
      if (cj < K) dots[(long)m * K + cj] = acc[j][r];
    }
  }
}

// ---------------------------------------------------------------------------
// Weight/codebook pre-conversion (memory-bound, once per launch)
// ---------------------------------------------------------------------------
__global__ void w_reorder_bf16_kernel(const float* __restrict__ w, __bf16* __restrict__ wr,
                                      int Cout, int Cin, int KV) {
  const long total = (long)Cout * Cin * KV;
  const long i = (long)blockIdx.x * blockDim.x + threadIdx.x;
  if (i >= total) return;
  const int ci = i % Cin; long t = i / Cin;
  const int tap = t % KV; const int co = t / KV;
  wr[i] = f2bf(w[((long)co * Cin + ci) * KV + tap]);  // [Co][Ci][tap] -> [Co][tap][Ci]
}

// first-conv weights: [64][3][64] fp32 -> padded [64][64][4] bf16 (cin==3 -> 0)
__global__ void w_first_pad_kernel(const float* __restrict__ w, __bf16* __restrict__ wr) {
  const int i = blockIdx.x * blockDim.x + threadIdx.x;
  if (i >= 64 * 64 * 4) return;
  const int ci = i & 3;
  const int tap = (i >> 2) & 63;
  const int co = i >> 8;
  wr[i] = (ci < 3) ? f2bf(w[(co * 3 + ci) * 64 + tap]) : f2bf(0.f);
}

__global__ void bf16_convert_kernel(const float* __restrict__ in, __bf16* __restrict__ out,
                                    long n) {
  const long i = (long)blockIdx.x * blockDim.x + threadIdx.x;
  if (i < n) out[i] = f2bf(in[i]);
}

// ---------------------------------------------------------------------------
// BatchNorm (training-mode batch stats) + apply (fused residual/relu)
// ---------------------------------------------------------------------------
__global__ void bn_stats_kernel(const float* __restrict__ x, float* __restrict__ ms,
                                int N, int C, long spatial) {
  const int c = blockIdx.x;
  float s = 0.f, s2 = 0.f;
  for (int n = 0; n < N; ++n) {
    const float* xp = x + ((long)n * C + c) * spatial;
    for (long sp = threadIdx.x; sp < spatial; sp += blockDim.x) {
      const float v = xp[sp];
      s += v; s2 += v * v;
    }
  }
  __shared__ float sh[256], sh2[256];
  sh[threadIdx.x] = s; sh2[threadIdx.x] = s2; __syncthreads();
  for (int st = 128; st > 0; st >>= 1) {
    if ((int)threadIdx.x < st) { sh[threadIdx.x] += sh[threadIdx.x + st];
                                 sh2[threadIdx.x] += sh2[threadIdx.x + st]; }
    __syncthreads();
  }
  if (threadIdx.x == 0) {
    const float cnt = (float)((long)N * spatial);
    const float m = sh[0] / cnt;
    ms[2 * c] = m;
    ms[2 * c + 1] = sh2[0] / cnt - m * m;
  }
}

// grid: (ceil(spatial/256), N*C); channel index = one scalar % per block.
__global__ void bn_apply_kernel(float* __restrict__ buf, const float* __restrict__ ms,
                                const float* __restrict__ g, const float* __restrict__ b,
                                const float* __restrict__ res,
                                int C, long spatial, int relu) {
  const int nc = blockIdx.y;
  const int c  = nc % C;
  const long sp = (long)blockIdx.x * blockDim.x + threadIdx.x;
  if (sp >= spatial) return;
  const long i = (long)nc * spatial + sp;
  const float m = ms[2 * c], v = ms[2 * c + 1];
  float o = (buf[i] - m) * rsqrtf(v + 1e-5f) * g[c] + b[c];
  if (res) o += res[i];
  if (relu) o = fmaxf(o, 0.f);
  buf[i] = o;
}

// ---------------------------------------------------------------------------
// VQ support kernels
// ---------------------------------------------------------------------------
__global__ void enorm_kernel(const float* __restrict__ emb, float* __restrict__ en,
                             int K, int C) {
  const int k = blockIdx.x * blockDim.x + threadIdx.x;
  if (k >= K) return;
  float s = 0.f;
  for (int c = 0; c < C; ++c) { const float v = emb[(long)k * C + c]; s += v * v; }
  en[k] = s;
}

__global__ void argmin_kernel(const float* __restrict__ dots, const float* __restrict__ en,
                              int* __restrict__ idx, int nrows, int K) {
  const int r = blockIdx.x * blockDim.x + threadIdx.x;
  if (r >= nrows) return;
  float best = 3.4e38f; int bi = 0;
  for (int k = 0; k < K; ++k) {
    const float d = en[k] - 2.f * dots[(long)r * K + k];
    if (d < best) { best = d; bi = k; }
  }
  idx[r] = bi;
}

__global__ void vq_gather_kernel(const float* __restrict__ x, const float* __restrict__ emb,
                                 const int* __restrict__ idx, float* __restrict__ q,
                                 float* __restrict__ acc,
                                 int B, int C, int D, int H, int W) {
  const long total = (long)B * C * D * H * W;
  const long i = (long)blockIdx.x * blockDim.x + threadIdx.x;
  float d2 = 0.f;
  if (i < total) {
    long w = i % W; long t = i / W;
    long h = t % H; t /= H;
    long d = t % D; t /= D;
    long c = t % C; long b = t / C;
    const long row = ((b * D + d) * H + h) * W + w;
    const float qv = emb[(long)idx[row] * C + c];
    const float xv = x[i];
    q[i] = qv;
    const float df = qv - xv;
    d2 = df * df;
  }
  __shared__ float sh[256];
  sh[threadIdx.x] = d2; __syncthreads();
  for (int s = 128; s > 0; s >>= 1) {
    if ((int)threadIdx.x < s) sh[threadIdx.x] += sh[threadIdx.x + s];
    __syncthreads();
  }
  if (threadIdx.x == 0) atomicAdd(acc, sh[0]);
}

__global__ void hist_kernel(const int* __restrict__ idx, int n, int* __restrict__ hist) {
  const int i = blockIdx.x * blockDim.x + threadIdx.x;
  if (i < n) atomicAdd(&hist[idx[i]], 1);
}

// ---------------------------------------------------------------------------
// Trilinear upsample x2 (half-pixel centers) into channel slice of concat buf
// ---------------------------------------------------------------------------
__global__ void upsample_tri_kernel(const float* __restrict__ in, float* __restrict__ out,
                                    int B, int C, int Di, int Hi, int Wi,
                                    int Ctot, int cOff) {
  const int Do = Di * 2, Ho = Hi * 2, Wo = Wi * 2;
  const long total = (long)B * C * Do * Ho * Wo;
  const long i = (long)blockIdx.x * blockDim.x + threadIdx.x;
  if (i >= total) return;
  int wo = i % Wo; long t = i / Wo;
  int ho = t % Ho; t /= Ho;
  int dd = t % Do; t /= Do;
  int c  = t % C;  int b = t / C;

  float sd = fminf(fmaxf((dd + 0.5f) * 0.5f - 0.5f, 0.f), (float)(Di - 1));
  float sh = fminf(fmaxf((ho + 0.5f) * 0.5f - 0.5f, 0.f), (float)(Hi - 1));
  float sw = fminf(fmaxf((wo + 0.5f) * 0.5f - 0.5f, 0.f), (float)(Wi - 1));
  const int d0 = (int)sd, h0 = (int)sh, w0 = (int)sw;
  const int d1 = d0 + 1 < Di ? d0 + 1 : Di - 1;
  const int h1 = h0 + 1 < Hi ? h0 + 1 : Hi - 1;
  const int w1 = w0 + 1 < Wi ? w0 + 1 : Wi - 1;
  const float fd = sd - d0, fh = sh - h0, fw = sw - w0;

  const long base = ((long)b * C + c) * Di;
  const long hw = (long)Hi * Wi;
#define AT(dz, hy, wx) in[(base + (dz)) * hw + (long)(hy) * Wi + (wx)]
  const float v =
      AT(d0, h0, w0) * (1 - fd) * (1 - fh) * (1 - fw) +
      AT(d0, h0, w1) * (1 - fd) * (1 - fh) * fw +
      AT(d0, h1, w0) * (1 - fd) * fh * (1 - fw) +
      AT(d0, h1, w1) * (1 - fd) * fh * fw +
      AT(d1, h0, w0) * fd * (1 - fh) * (1 - fw) +
      AT(d1, h0, w1) * fd * (1 - fh) * fw +
      AT(d1, h1, w0) * fd * fh * (1 - fw) +
      AT(d1, h1, w1) * fd * fh * fw;
#undef AT
  out[(((long)b * Ctot + cOff + c) * Do + dd) * (long)Ho * Wo + (long)ho * Wo + wo] = v;
}

__global__ void copy_channels_kernel(const float* __restrict__ in, float* __restrict__ out,
                                     int B, int C, long spatial, int Ctot, int cOff) {
  const long total = (long)B * C * spatial;
  const long i = (long)blockIdx.x * blockDim.x + threadIdx.x;
  if (i >= total) return;
  const long sp = i % spatial; long t = i / spatial;
  const long c = t % C; const long b = t / C;
  out[((b * Ctot + cOff + c) * spatial) + sp] = in[i];
}

// ---------------------------------------------------------------------------
// Final scalars: total_loss, avg_ent, ent_top, ent_bot, used_top, used_bot
// ---------------------------------------------------------------------------
__global__ void finalize_kernel(const int* __restrict__ histT, const int* __restrict__ histB,
                                const float* __restrict__ lacc, float* __restrict__ out) {
  __shared__ float se[512];
  __shared__ int   su[512];
  __shared__ float resE; __shared__ int resU;
  const int tid = threadIdx.x;

  float e = 0.f; int u = 0;
  if (tid < 128) {
    const int h = histT[tid];
    if (h > 0) { const float p = (float)h / 2048.f; e = -p * log2f(p); u = 1; }
  }
  se[tid] = e; su[tid] = u; __syncthreads();
  for (int s = 256; s > 0; s >>= 1) {
    if (tid < s) { se[tid] += se[tid + s]; su[tid] += su[tid + s]; }
    __syncthreads();
  }
  if (tid == 0) { resE = se[0]; resU = su[0]; }
  __syncthreads();

  e = 0.f; u = 0;
  {
    const int h = histB[tid];
    if (h > 0) { const float p = (float)h / 16384.f; e = -p * log2f(p); u = 1; }
  }
  se[tid] = e; su[tid] = u; __syncthreads();
  for (int s = 256; s > 0; s >>= 1) {
    if (tid < s) { se[tid] += se[tid + s]; su[tid] += su[tid + s]; }
    __syncthreads();
  }
  if (tid == 0) {
    const float entB = se[0]; const int usedB = su[0];
    const float entT = resE;  const int usedT = resU;
    const float lossT = lacc[0] / (2048.f * 512.f) * 1.01f;
    const float lossB = lacc[1] / (16384.f * 256.f) * 1.05f;
    const bool both = (entT > 0.f) && (entB > 0.f);
    const float avg = both ? 0.5f * (entT + entB) : -1.0f;
    const float entLoss = both ? -avg * 0.01f : 0.f;
    out[0] = lossT + lossB + entLoss;
    out[1] = avg;
    out[2] = entT;
    out[3] = entB;
    out[4] = (float)usedT;
    out[5] = (float)usedB;
  }
}

// ---------------------------------------------------------------------------
// Host orchestration
// ---------------------------------------------------------------------------
#define LAUNCH_FCONV(KS, STR, PAD, LD, CIN, DI, HI, WI, NT, X_, WR_, B_, Y_, NN, CO, DO_, HO_, WO_, ACT_) \
  do {                                                                                                    \
    long M_ = (long)(NN) * (DO_) * (HO_) * (WO_);                                                         \
    dim3 g_((unsigned)((M_ + 15) / 16), (unsigned)(((CO) + 16 * (NT) - 1) / (16 * (NT))), 1);             \
    conv3d_wmma_fast<KS, STR, PAD, LD, CIN, DI, HI, WI, NT>                                               \
        <<<g_, 32, 0, stream>>>(X_, WR_, B_, Y_, NN, CO, DO_, HO_, WO_, ACT_);                            \
  } while (0)

extern "C" void kernel_launch(void* const* d_in, const int* in_sizes, int n_in,
                              void* d_out, int out_size, void* d_ws, size_t ws_size,
                              hipStream_t stream) {
  (void)in_sizes; (void)n_in; (void)out_size; (void)ws_size;
  const float* P[43];
  for (int i = 0; i < 43; ++i) P[i] = (const float*)d_in[i];
  const float* X       = P[0];
  const float* eb_w1   = P[1];  const float* eb_b1  = P[2];
  const float* r64w1   = P[3];  const float* r64b1  = P[4];
  const float* r64g1   = P[5];  const float* r64bb1 = P[6];
  const float* r64w2   = P[7];  const float* r64b2  = P[8];
  const float* r64g2   = P[9];  const float* r64bb2 = P[10];
  const float* eb_w2   = P[11]; const float* eb_b2  = P[12];
  const float* r256w1  = P[13]; const float* r256b1 = P[14];
  const float* r256g1  = P[15]; const float* r256bb1= P[16];
  const float* r256w2  = P[17]; const float* r256b2 = P[18];
  const float* r256g2  = P[19]; const float* r256bb2= P[20];
  const float* et_w    = P[21]; const float* et_b   = P[22];
  const float* r512w1  = P[23]; const float* r512b1 = P[24];
  const float* r512g1  = P[25]; const float* r512bb1= P[26];
  const float* r512w2  = P[27]; const float* r512b2 = P[28];
  const float* r512g2  = P[29]; const float* r512bb2= P[30];
  const float* embT    = P[31]; const float* embB   = P[32];
  const float* d_w1    = P[33]; const float* d_b1   = P[34];
  const float* d_g1    = P[35]; const float* d_bb1  = P[36];
  const float* d_w2    = P[37]; const float* d_b2   = P[38];
  const float* d_g2    = P[39]; const float* d_bb2  = P[40];
  const float* d_w3    = P[41]; const float* d_b3   = P[42];

  // workspace bump allocator (floats, 256B aligned chunks)
  float* wsf = (float*)d_ws;
  size_t off = 0;
  auto alloc   = [&](size_t n) { float* p = wsf + off; off += (n + 63) & ~(size_t)63; return p; };
  auto allocbf = [&](size_t n) { return (__bf16*)alloc((n + 1) / 2); };

  float* big0  = alloc(16777216);   // zb1 then d1
  float* big1  = alloc(8388608);
  float* big2  = alloc(8388608);
  float* zb    = alloc(4194304);
  float* zt    = alloc(1048576);
  float* zt2   = alloc(1048576);
  float* dots  = alloc(8388608);
  float* qtop  = alloc(1048576);
  float* qbot  = alloc(4194304);
  float* zc    = alloc(12582912);
  float* d2b   = alloc(134217728);
  float* stats = alloc(1024);
  float* en    = alloc(512);
  float* lacc  = alloc(8);
  int* idxT  = (int*)alloc(2048);
  int* idxB  = (int*)alloc(16384);
  int* histT = (int*)alloc(128);
  int* histB = (int*)alloc(512);

  // bf16 reordered weights [Cout][tap][Cin] + padded first conv + codebooks
  __bf16* wrEb1 = allocbf(64 * 64 * 4);
  __bf16* wr64a = allocbf(64 * 64 * 27);
  __bf16* wr64b = allocbf(64 * 64 * 27);
  __bf16* wrEb2 = allocbf(256 * 64 * 64);
  __bf16* wr256a = allocbf(256 * 256 * 27);
  __bf16* wr256b = allocbf(256 * 256 * 27);
  __bf16* wrEt  = allocbf(512 * 256 * 64);
  __bf16* wr512a = allocbf(512 * 512 * 27);
  __bf16* wr512b = allocbf(512 * 512 * 27);
  __bf16* wrD1  = allocbf(256 * 768 * 64);
  __bf16* wrD2  = allocbf(128 * 256 * 64);
  __bf16* wrD3  = allocbf(3 * 128 * 27);
  __bf16* ebfT  = allocbf(128 * 512);
  __bf16* ebfB  = allocbf(512 * 256);

  hipMemsetAsync(lacc, 0, 8 * sizeof(float), stream);
  hipMemsetAsync(histT, 0, 128 * sizeof(int), stream);
  hipMemsetAsync(histB, 0, 512 * sizeof(int), stream);

  // ---------------- weight pre-conversion (memory-bound) ----------------
  auto reorder = [&](const float* w, __bf16* wr, int Cout, int Cin, int KV) {
    const long total = (long)Cout * Cin * KV;
    w_reorder_bf16_kernel<<<(unsigned)((total + 255) / 256), 256, 0, stream>>>(w, wr,
                                                                               Cout, Cin, KV);
  };
  w_first_pad_kernel<<<64, 256, 0, stream>>>(eb_w1, wrEb1);
  reorder(r64w1, wr64a, 64, 64, 27);
  reorder(r64w2, wr64b, 64, 64, 27);
  reorder(eb_w2, wrEb2, 256, 64, 64);
  reorder(r256w1, wr256a, 256, 256, 27);
  reorder(r256w2, wr256b, 256, 256, 27);
  reorder(et_w, wrEt, 512, 256, 64);
  reorder(r512w1, wr512a, 512, 512, 27);
  reorder(r512w2, wr512b, 512, 512, 27);
  reorder(d_w1, wrD1, 256, 768, 64);
  reorder(d_w2, wrD2, 128, 256, 64);
  reorder(d_w3, wrD3, 3, 128, 27);
  bf16_convert_kernel<<<(65536 + 255) / 256, 256, 0, stream>>>(embT, ebfT, 65536);
  bf16_convert_kernel<<<(131072 + 255) / 256, 256, 0, stream>>>(embB, ebfB, 131072);

  auto batchnorm = [&](float* buf, const float* g, const float* b, const float* res,
                       int relu, int N, int C, long spatial) {
    bn_stats_kernel<<<C, 256, 0, stream>>>(buf, stats, N, C, spatial);
    dim3 g2((unsigned)((spatial + 255) / 256), (unsigned)(N * C), 1);
    bn_apply_kernel<<<g2, 256, 0, stream>>>(buf, stats, g, b, res, C, spatial, relu);
  };

  // ---------------- Encoder bottom ----------------
  conv3d_wmma_first<<<dim3(8192), 32, 0, stream>>>(X, wrEb1, eb_b1, big0);
  // rb64
  LAUNCH_FCONV(3,1,1,1,64, 8,64,64, 4, big0, wr64a, r64b1, big1, 4, 64, 8,64,64, ACT_NONE);
  batchnorm(big1, r64g1, r64bb1, nullptr, 1, 4, 64, (long)8 * 64 * 64);
  LAUNCH_FCONV(3,1,1,1,64, 8,64,64, 4, big1, wr64b, r64b2, big2, 4, 64, 8,64,64, ACT_NONE);
  batchnorm(big2, r64g2, r64bb2, big0, 1, 4, 64, (long)8 * 64 * 64);
  // eb_w2
  LAUNCH_FCONV(4,2,1,1,64, 8,64,64, 4, big2, wrEb2, eb_b2, zb, 4, 256, 4,32,32, ACT_RELU);
  // rb256
  LAUNCH_FCONV(3,1,1,1,256, 4,32,32, 4, zb, wr256a, r256b1, big1, 4, 256, 4,32,32, ACT_NONE);
  batchnorm(big1, r256g1, r256bb1, nullptr, 1, 4, 256, (long)4 * 32 * 32);
  LAUNCH_FCONV(3,1,1,1,256, 4,32,32, 4, big1, wr256b, r256b2, big2, 4, 256, 4,32,32, ACT_NONE);
  batchnorm(big2, r256g2, r256bb2, zb, 1, 4, 256, (long)4 * 32 * 32);   // zbF -> big2

  // ---------------- Encoder top ----------------
  LAUNCH_FCONV(4,2,1,1,256, 4,32,32, 4, big2, wrEt, et_b, zt, 4, 512, 2,16,16, ACT_RELU);
  LAUNCH_FCONV(3,1,1,1,512, 2,16,16, 4, zt, wr512a, r512b1, big1, 4, 512, 2,16,16, ACT_NONE);
  batchnorm(big1, r512g1, r512bb1, nullptr, 1, 4, 512, (long)2 * 16 * 16);
  LAUNCH_FCONV(3,1,1,1,512, 2,16,16, 4, big1, wr512b, r512b2, zt2, 4, 512, 2,16,16, ACT_NONE);
  batchnorm(zt2, r512g2, r512bb2, zt, 1, 4, 512, (long)2 * 16 * 16);    // ztF -> zt2

  // ---------------- VQ top: 2048 rows, C=512, K=128 ----------------
  enorm_kernel<<<2, 64, 0, stream>>>(embT, en, 128, 512);
  vq_dots_fast<512, 4><<<dim3(128, 2), 32, 0, stream>>>(zt2, ebfT, dots, 4, 2, 16, 16, 128);
  argmin_kernel<<<(2048 + 255) / 256, 256, 0, stream>>>(dots, en, idxT, 2048, 128);
  vq_gather_kernel<<<(1048576 + 255) / 256, 256, 0, stream>>>(zt2, embT, idxT, qtop,
                                                              &lacc[0], 4, 512, 2, 16, 16);
  hist_kernel<<<(2048 + 255) / 256, 256, 0, stream>>>(idxT, 2048, histT);

  // ---------------- VQ bottom: 16384 rows, C=256, K=512 ----------------
  enorm_kernel<<<8, 64, 0, stream>>>(embB, en, 512, 256);
  vq_dots_fast<256, 4><<<dim3(1024, 8), 32, 0, stream>>>(big2, ebfB, dots, 4, 4, 32, 32, 512);
  argmin_kernel<<<(16384 + 255) / 256, 256, 0, stream>>>(dots, en, idxB, 16384, 512);
  vq_gather_kernel<<<(4194304 + 255) / 256, 256, 0, stream>>>(big2, embB, idxB, qbot,
                                                              &lacc[1], 4, 256, 4, 32, 32);
  hist_kernel<<<(16384 + 255) / 256, 256, 0, stream>>>(idxB, 16384, histB);

  // ---------------- Concat ----------------
  upsample_tri_kernel<<<(8388608 + 255) / 256, 256, 0, stream>>>(qtop, zc,
                                                                 4, 512, 2, 16, 16, 768, 0);
  copy_channels_kernel<<<(4194304 + 255) / 256, 256, 0, stream>>>(qbot, zc,
                                                                  4, 256, (long)4 * 32 * 32,
                                                                  768, 512);

  // ---------------- Decoder (ConvT as lhs_dilation-2 conv, pad k-1-p=2) ----------------
  LAUNCH_FCONV(4,1,2,2,768, 4,32,32, 4, zc, wrD1, d_b1, big0, 4, 256, 8,64,64, ACT_NONE);
  batchnorm(big0, d_g1, d_bb1, nullptr, 1, 4, 256, (long)8 * 64 * 64);
  LAUNCH_FCONV(4,1,2,2,256, 8,64,64, 4, big0, wrD2, d_b2, d2b, 4, 128, 16,128,128, ACT_NONE);
  batchnorm(d2b, d_g2, d_bb2, nullptr, 1, 4, 128, (long)16 * 128 * 128);
  LAUNCH_FCONV(3,1,1,1,128, 16,128,128, 1, d2b, wrD3, d_b3, (float*)d_out, 4, 3,
               16,128,128, ACT_SIGM);

  // ---------------- Scalars ----------------
  finalize_kernel<<<1, 512, 0, stream>>>(histT, histB, lacc, (float*)d_out + 3145728);
}
